// NawrotGemmaMiniBitterLLM_75453985456506
// MI455X (gfx1250) — compile-verified, matrix operand order checked by reading, not run on hardware
//
#include <hip/hip_runtime.h>

// ---------------- model constants ----------------
constexpr int kD   = 1024;
constexpr int kH   = 16;
constexpr int kHD  = 64;      // kD / kH
constexpr int kV   = 32000;
constexpr int kB   = 2;
constexpr int kT   = 1024;
constexpr int kS1  = kT + 1;  // mid sequence length
constexpr int kWIN = 64;
constexpr float kSOFTCAP = 50.0f;
constexpr float kSCALE   = 0.125f;   // 64^-0.5
constexpr float kEPS     = 1e-6f;

typedef __attribute__((ext_vector_type(16))) _Float16 v16h;
typedef __attribute__((ext_vector_type(8)))  float    v8f;

#define GF_BIAS 1
#define GF_RELU 2

// ---------------- WMMA GEMM: C[M,N] = A[M,K] @ W[K,N] (+bias)(+relu) ---------
// Block = 128 threads = 4 waves; block tile 64(M) x 64(N); K-step 32.
// A/B tiles staged f32 in LDS via GLOBAL_LOAD_ASYNC_TO_LDS_B128 (ASYNCcnt),
// double-buffered; each wave computes 16x64 (4 WMMA accumulators).
// Fragment layouts per CDNA5 ISA 7.12.2:
//   A (16-bit 16x32): lane m=l%16; khalf=(l>=16)?8:0; half e of vgpr v holds
//     K = (v>>2)*16 + khalf + (v&3)*2 + e;  B mirrors with N=l%16.
//   C/D: vgpr r -> M = r + ((l>=16)?8:0), N = l%16.
__global__ void k_gemm(const float* __restrict__ A, const float* __restrict__ Bw,
                       const float* __restrict__ bias, float* __restrict__ C,
                       int M, int N, int K, int flags) {
  __shared__ float As[2][64 * 32];   // [m][k], k-contiguous
  __shared__ float Bs[2][32 * 64];   // [k][n], n-contiguous

  const int tid  = threadIdx.x;
  const int lane = tid & 31;
  const int wave = tid >> 5;
  const int tm   = blockIdx.y;              // 64-row M tile
  const int bn0  = blockIdx.x * 64;         // N base (N % 64 == 0 always)

  // staging assignment (wave-uniform loop trip counts, per-lane addresses)
  const int arow = tid >> 3;                // 16 rows per issue
  const int acol = (tid & 7) * 4;           // 4 floats = 16B per lane
  const int brow = tid >> 4;                // 8 k-rows per issue
  const int bcol = (tid & 15) * 4;

  auto stage = [&](int buf, int k0) {
    float* asb = &As[buf][0];
    float* bsb = &Bs[buf][0];
#pragma unroll
    for (int a = 0; a < 4; ++a) {
      const int r = a * 16 + arow;
      int gm = tm * 64 + r; if (gm >= M) gm = M - 1;          // clamp (EXEC stays full)
      const float* gsrc = A + (size_t)gm * K + k0 + acol;
      const unsigned ldst = (unsigned)(unsigned long long)(void*)(asb + r * 32 + acol);
      asm volatile("global_load_async_to_lds_b128 %0, %1, off"
                   :: "v"(ldst), "v"((unsigned long long)gsrc) : "memory");
    }
#pragma unroll
    for (int b = 0; b < 4; ++b) {
      const int kr = b * 8 + brow;
      const float* gsrc = Bw + (size_t)(k0 + kr) * N + bn0 + bcol;
      const unsigned ldst = (unsigned)(unsigned long long)(void*)(bsb + kr * 64 + bcol);
      asm volatile("global_load_async_to_lds_b128 %0, %1, off"
                   :: "v"(ldst), "v"((unsigned long long)gsrc) : "memory");
    }
  };

  const int ln    = lane & 15;
  const int khalf = (lane & 16) ? 8 : 0;
  const int arow_c = wave * 16 + ln;        // this wave's A row in the tile

  v8f acc[4] = {};

  stage(0, 0);
  const int nsteps = K / 32;                // K % 32 == 0 for all our GEMMs
  for (int s = 0; s < nsteps; ++s) {
    const int cur = s & 1;
    const bool more = (s + 1) < nsteps;
    if (more) {
      stage(cur ^ 1, (s + 1) * 32);
      asm volatile("s_wait_asynccnt 0x8" ::: "memory");   // stage s complete
    } else {
      asm volatile("s_wait_asynccnt 0x0" ::: "memory");
    }
    __syncthreads();

    const float* as = &As[cur][0];
    const float* bs = &Bs[cur][0];
    v16h af;
#pragma unroll
    for (int vv = 0; vv < 8; ++vv) {
      const int kb = ((vv >> 2) << 4) + khalf + ((vv & 3) << 1);
      af[2 * vv]     = (_Float16)as[arow_c * 32 + kb];
      af[2 * vv + 1] = (_Float16)as[arow_c * 32 + kb + 1];
    }
#pragma unroll
    for (int n0 = 0; n0 < 4; ++n0) {
      v16h bf;
#pragma unroll
      for (int vv = 0; vv < 8; ++vv) {
        const int kb = ((vv >> 2) << 4) + khalf + ((vv & 3) << 1);
        bf[2 * vv]     = (_Float16)bs[kb * 64 + n0 * 16 + ln];
        bf[2 * vv + 1] = (_Float16)bs[(kb + 1) * 64 + n0 * 16 + ln];
      }
      acc[n0] = __builtin_amdgcn_wmma_f32_16x16x32_f16(false, af, false, bf,
                                                       (short)0, acc[n0], false, false);
    }
    __syncthreads();   // all waves done with buffer `cur` before it is restaged
  }

  const int mrow = tm * 64 + wave * 16 + ((lane & 16) ? 8 : 0);
#pragma unroll
  for (int n0 = 0; n0 < 4; ++n0) {
    const int cn = bn0 + n0 * 16 + ln;
#pragma unroll
    for (int r = 0; r < 8; ++r) {
      const int cm = mrow + r;
      if (cm < M) {
        float v = acc[n0][r];
        if (flags & GF_BIAS) v += bias[cn];
        if (flags & GF_RELU) v = fmaxf(v, 0.f);
        C[(size_t)cm * N + cn] = v;
      }
    }
  }
}

// ---------------- RMSNorm: out = (res?) + x*rsqrt(mean(x^2)+eps)*(1+w) -------
__global__ void k_rmsnorm(const float* __restrict__ in, const float* __restrict__ w,
                          const float* __restrict__ res, float* __restrict__ out) {
  __shared__ float red[256];
  const int row = blockIdx.x, tid = threadIdx.x;
  const float* x = in + (size_t)row * kD;
  float s = 0.f;
  for (int i = tid; i < kD; i += 256) { float t = x[i]; s += t * t; }
  red[tid] = s; __syncthreads();
  for (int st = 128; st > 0; st >>= 1) { if (tid < st) red[tid] += red[tid + st]; __syncthreads(); }
  const float inv = rsqrtf(red[0] / kD + kEPS);
  for (int i = tid; i < kD; i += 256) {
    const float nv = x[i] * inv * (1.f + w[i]);
    out[(size_t)row * kD + i] = (res ? res[(size_t)row * kD + i] : 0.f) + nv;
  }
}

// ---------------- RoPE (half-rotate form), in-place on q and k ---------------
__global__ void k_rope(float* __restrict__ q, float* __restrict__ k,
                       const float* __restrict__ pos, int S) {
  const int s = blockIdx.x, b = blockIdx.y;
  const int h = threadIdx.x >> 5;   // head 0..15
  const int i = threadIdx.x & 31;   // freq index 0..31
  const float p  = pos ? pos[b * S + s] : (float)s;
  const float ang = p * __powf(10000.f, -(float)i / 32.f);
  const float c = __cosf(ang), sn = __sinf(ang);
  const size_t idx = ((size_t)(b * S + s)) * kD + h * kHD + i;
  const float q1 = q[idx], q2 = q[idx + 32];
  q[idx]      = q1 * c - q2 * sn;
  q[idx + 32] = q2 * c + q1 * sn;
  const float k1 = k[idx], k2 = k[idx + 32];
  k[idx]      = k1 * c - k2 * sn;
  k[idx + 32] = k2 * c + k1 * sn;
}

// ---------------- attention: one block (64 threads) per (b,h,query) ---------
// online softmax over causal (optionally windowed) keys; tanh softcap.
__global__ void k_attn(const float* __restrict__ q, const float* __restrict__ k,
                       const float* __restrict__ v, float* __restrict__ o,
                       int S, int window) {
  __shared__ float sq[64];
  __shared__ float sw[64];
  __shared__ float red[64];
  const int tid = threadIdx.x;
  const int tq  = blockIdx.x;
  const int bh  = blockIdx.y;
  const int b = bh / kH, h = bh % kH;
  const size_t base = ((size_t)b * S) * kD + (size_t)h * kHD;
  sq[tid] = q[base + (size_t)tq * kD + tid];
  __syncthreads();

  float m = -1e30f, l = 0.f, acc = 0.f;
  int kstart = tq - window + 1; if (kstart < 0) kstart = 0;
  for (int c0 = kstart; c0 <= tq; c0 += 64) {
    const int kk = c0 + tid;
    float s = -1e30f;
    if (kk <= tq) {
      const float* krow = k + base + (size_t)kk * kD;
      float d = 0.f;
#pragma unroll 8
      for (int i = 0; i < kHD; ++i) d += sq[i] * krow[i];
      d *= kSCALE;
      s = kSOFTCAP * tanhf(d / kSOFTCAP);
    }
    red[tid] = s; __syncthreads();
    for (int st = 32; st > 0; st >>= 1) { if (tid < st) red[tid] = fmaxf(red[tid], red[tid + st]); __syncthreads(); }
    const float mnew = fmaxf(m, red[0]);
    __syncthreads();
    const float p = (kk <= tq) ? __expf(s - mnew) : 0.f;
    sw[tid] = p;
    red[tid] = p; __syncthreads();
    for (int st = 32; st > 0; st >>= 1) { if (tid < st) red[tid] += red[tid + st]; __syncthreads(); }
    const float scale = __expf(m - mnew);
    l = l * scale + red[0];
    acc *= scale;
    int lim = tq - c0; if (lim > 63) lim = 63;
    for (int j = 0; j <= lim; ++j)
      acc += sw[j] * v[base + (size_t)(c0 + j) * kD + tid];
    m = mnew;
    __syncthreads();
  }
  o[base + (size_t)tq * kD + tid] = acc / (l + 1e-20f);
}

// ---------------- embedding gather --------------------------------------
__global__ void k_embed(const int* __restrict__ ids, const float* __restrict__ emb,
                        float* __restrict__ x) {
  const int row = blockIdx.x, tid = threadIdx.x;
  const int id = ids[row];
  for (int i = tid; i < kD; i += 256)
    x[(size_t)row * kD + i] = emb[(size_t)id * kD + i];
}

// ---------------- gelu(tanh) * u, in place on g -------------------------
__global__ void k_glu(float* __restrict__ g, const float* __restrict__ u, size_t n) {
  const size_t i = (size_t)blockIdx.x * blockDim.x + threadIdx.x;
  if (i < n) {
    const float x = g[i];
    const float t = 0.7978845608028654f * (x + 0.044715f * x * x * x);
    g[i] = 0.5f * x * (1.f + tanhf(t)) * u[i];
  }
}

// ---------------- boundary predictor tail: dot + gumbel-sigmoid ---------
__global__ void k_bp_final(const float* __restrict__ h1, const float* __restrict__ w2,
                           const float* __restrict__ b2,
                           float* __restrict__ soft_out, float* __restrict__ hard_out,
                           float* __restrict__ soft_ws, float* __restrict__ hard_ws) {
  __shared__ float red[256];
  const int row = blockIdx.x, tid = threadIdx.x;
  float s = 0.f;
  for (int i = tid; i < kD; i += 256) s += h1[(size_t)row * kD + i] * w2[i];
  red[tid] = s; __syncthreads();
  for (int st = 128; st > 0; st >>= 1) { if (tid < st) red[tid] += red[tid + st]; __syncthreads(); }
  if (tid == 0) {
    const float logit = red[0] + b2[0];
    unsigned int hs = (unsigned)row * 747796405u + 2891336453u;
    hs ^= hs >> 17; hs *= 0xed5ad4bbu; hs ^= hs >> 11; hs *= 0xac4c1b51u; hs ^= hs >> 15;
    const float u = ((float)hs * (1.0f / 4294967296.0f)) * (1.f - 2e-6f) + 1e-6f;
    const float g = __logf(u) - log1pf(-u);
    const float sft = 1.f / (1.f + __expf(-(logit + g)));
    const float hrd = sft > 0.5f ? 1.f : 0.f;
    soft_out[row] = sft; hard_out[row] = hrd;
    soft_ws[row] = sft;  hard_ws[row] = hrd;
  }
}

// ---------------- boundary cumsum + segment stats (per batch) -----------
__global__ void k_seg(const float* __restrict__ hard, int* __restrict__ seg_ex,
                      int* __restrict__ seg_in, float* __restrict__ cnt,
                      float* __restrict__ possum, float* __restrict__ pos_ds) {
  __shared__ int segs[kT];
  const int b = blockIdx.x, tid = threadIdx.x;
  for (int s = tid; s < kT; s += 256) { cnt[b * kT + s] = 0.f; possum[b * kT + s] = 0.f; }
  __syncthreads();
  if (tid == 0) {
    int c = 0;
    for (int t = 0; t < kT; ++t) {
      segs[t] = c;                                   // exclusive cumsum
      c += (hard[b * kT + t] > 0.5f) ? 1 : 0;
      seg_in[b * kT + t] = c;                        // inclusive cumsum
    }
  }
  __syncthreads();
  for (int t = tid; t < kT; t += 256) {
    const int s = segs[t];
    seg_ex[b * kT + t] = s;
    atomicAdd(&cnt[b * kT + s], 1.f);
    atomicAdd(&possum[b * kT + s], (float)t);
  }
  __syncthreads();
  if (tid == 0) pos_ds[b * kS1] = 0.f;
  for (int s = tid; s < kT; s += 256)
    pos_ds[b * kS1 + s + 1] = possum[b * kT + s] / (cnt[b * kT + s] + 1e-9f);
}

// ---------------- downsample init / accumulate --------------------------
__global__ void k_ds_init(const float* __restrict__ null_g, float* __restrict__ x_ds) {
  const int s = blockIdx.x, b = blockIdx.y, tid = threadIdx.x;
  for (int i = tid; i < kD; i += 256)
    x_ds[((size_t)(b * kS1 + s)) * kD + i] = (s == 0) ? null_g[i] : 0.f;
}

__global__ void k_ds_acc(const float* __restrict__ x, const int* __restrict__ seg_ex,
                         const float* __restrict__ cnt, float* __restrict__ x_ds) {
  const int t = blockIdx.x, b = blockIdx.y, tid = threadIdx.x;
  const int s = seg_ex[b * kT + t];
  const float invc = 1.f / (cnt[b * kT + s] + 1e-9f);
  for (int i = tid; i < kD; i += 256)
    atomicAdd(&x_ds[((size_t)(b * kS1 + s + 1)) * kD + i],
              x[((size_t)(b * kT + t)) * kD + i] * invc);
}

// ---------------- upsample + residual: x += (y_ds - x_ds)[seg_in] -------
__global__ void k_upsample(float* __restrict__ x, const float* __restrict__ y_ds,
                           const float* __restrict__ x_ds, const int* __restrict__ seg_in) {
  const int t = blockIdx.x, b = blockIdx.y, tid = threadIdx.x;
  const int idx = seg_in[b * kT + t];
  const size_t srow = ((size_t)(b * kS1 + idx)) * kD;
  const size_t drow = ((size_t)(b * kT + t)) * kD;
  for (int i = tid; i < kD; i += 256)
    x[drow + i] += y_ds[srow + i] - x_ds[srow + i];
}

// ---------------- row log-softmax (in place) ----------------------------
__global__ void k_logsoftmax(float* __restrict__ lg, int cols) {
  __shared__ float red[256];
  const int row = blockIdx.x, tid = threadIdx.x;
  float* p = lg + (size_t)row * cols;
  float mx = -1e30f;
  for (int i = tid; i < cols; i += 256) mx = fmaxf(mx, p[i]);
  red[tid] = mx; __syncthreads();
  for (int st = 128; st > 0; st >>= 1) { if (tid < st) red[tid] = fmaxf(red[tid], red[tid + st]); __syncthreads(); }
  const float rowmax = red[0]; __syncthreads();
  float s = 0.f;
  for (int i = tid; i < cols; i += 256) s += __expf(p[i] - rowmax);
  red[tid] = s; __syncthreads();
  for (int st = 128; st > 0; st >>= 1) { if (tid < st) red[tid] += red[tid + st]; __syncthreads(); }
  const float lse = rowmax + __logf(red[0]);
  for (int i = tid; i < cols; i += 256) p[i] -= lse;
}

// ======================== host orchestration ============================
extern "C" void kernel_launch(void* const* d_in, const int* in_sizes, int n_in,
                              void* d_out, int out_size, void* d_ws, size_t ws_size,
                              hipStream_t stream) {
  (void)in_sizes; (void)n_in; (void)out_size; (void)ws_size;

  const int*   input_ids = (const int*)d_in[0];
  const float* embedding = (const float*)d_in[1];
  // layers: 11 tensors each {wq,wk,wv,wo,wg,wu,wd,ln_in,ln_pa,ln_pf,ln_pff}
  const int LAYER_BASE[6] = {2, 13, 24, 35, 46, 57};
  const float* bp_w1 = (const float*)d_in[68];
  const float* bp_b1 = (const float*)d_in[69];
  const float* bp_w2 = (const float*)d_in[70];
  const float* bp_b2 = (const float*)d_in[71];
  const float* null_group = (const float*)d_in[72];
  const float* out_w = (const float*)d_in[73];
  const float* out_b = (const float*)d_in[74];

  // -------- workspace carve --------
  char* wp = (char*)d_ws; size_t off = 0;
  auto walloc = [&](size_t bytes) -> void* {
    void* p = wp + off; off += bytes; off = (off + 255) & ~(size_t)255; return p;
  };
  const size_t bufA = (size_t)kB * kS1 * kD * sizeof(float);        // padded row buffer
  const size_t bufI = (size_t)kB * kS1 * 4 * kD * sizeof(float);    // 4D intermediate
  float* xbuf = (float*)walloc((size_t)kB * kT * kD * sizeof(float));
  float* xn   = (float*)walloc(bufA);
  float* qb   = (float*)walloc(bufA);
  float* kb   = (float*)walloc(bufA);
  float* vb   = (float*)walloc(bufA);
  float* ob   = (float*)walloc(bufA);
  float* hbuf = (float*)walloc(bufA);
  float* gbuf = (float*)walloc(bufI);
  float* ubuf = (float*)walloc(bufI);
  float* xds  = (float*)walloc(bufA);
  float* yds  = (float*)walloc(bufA);
  int*   seg_ex = (int*)walloc((size_t)kB * kT * sizeof(int));
  int*   seg_in = (int*)walloc((size_t)kB * kT * sizeof(int));
  float* cnt    = (float*)walloc((size_t)kB * kT * sizeof(float));
  float* possum = (float*)walloc((size_t)kB * kT * sizeof(float));
  float* posds  = (float*)walloc((size_t)kB * kS1 * sizeof(float));
  float* soft_ws = (float*)walloc((size_t)kB * kT * sizeof(float));
  float* hard_ws = (float*)walloc((size_t)kB * kT * sizeof(float));

  float* out_logits = (float*)d_out;
  float* out_soft = out_logits + (size_t)kB * kT * kV;
  float* out_hard = out_soft + (size_t)kB * kT;

  auto LP = [&](int li, int idx) -> const float* {
    return (const float*)d_in[LAYER_BASE[li] + idx];
  };
  auto gemm = [&](const float* A, const float* W, const float* bias, float* C,
                  int M, int N, int K, int flags) {
    dim3 grid(N / 64, (M + 63) / 64);      // N % 64 == 0 for all GEMMs here
    k_gemm<<<grid, 128, 0, stream>>>(A, W, bias, C, M, N, K, flags);
  };
  auto run_layer = [&](int li, float* x, int S, int inter, const float* pos, int window) {
    const int M = kB * S;
    k_rmsnorm<<<M, 256, 0, stream>>>(x, LP(li, 7), nullptr, xn);
    gemm(xn, LP(li, 0), nullptr, qb, M, kD, kD, 0);
    gemm(xn, LP(li, 1), nullptr, kb, M, kD, kD, 0);
    gemm(xn, LP(li, 2), nullptr, vb, M, kD, kD, 0);
    k_rope<<<dim3(S, kB), 512, 0, stream>>>(qb, kb, pos, S);
    k_attn<<<dim3(S, kB * kH), 64, 0, stream>>>(qb, kb, vb, ob, S, window);
    gemm(ob, LP(li, 3), nullptr, hbuf, M, kD, kD, 0);
    k_rmsnorm<<<M, 256, 0, stream>>>(hbuf, LP(li, 8), x, x);       // x += rms(h)
    k_rmsnorm<<<M, 256, 0, stream>>>(x, LP(li, 9), nullptr, xn);
    gemm(xn, LP(li, 4), nullptr, gbuf, M, inter, kD, 0);
    gemm(xn, LP(li, 5), nullptr, ubuf, M, inter, kD, 0);
    const size_t n = (size_t)M * inter;
    k_glu<<<(unsigned)((n + 255) / 256), 256, 0, stream>>>(gbuf, ubuf, n);
    gemm(gbuf, LP(li, 6), nullptr, hbuf, M, kD, inter, 0);
    k_rmsnorm<<<M, 256, 0, stream>>>(hbuf, LP(li, 10), x, x);      // x += rms(h)
  };

  // -------- forward pass --------
  k_embed<<<kB * kT, 256, 0, stream>>>(input_ids, embedding, xbuf);

  run_layer(0, xbuf, kT, kD, nullptr, kWIN);      // down layers (windowed causal)
  run_layer(1, xbuf, kT, kD, nullptr, kWIN);

  // boundary predictor
  gemm(xbuf, bp_w1, bp_b1, gbuf, kB * kT, kD, kD, GF_BIAS | GF_RELU);
  k_bp_final<<<kB * kT, 256, 0, stream>>>(gbuf, bp_w2, bp_b2,
                                          out_soft, out_hard, soft_ws, hard_ws);
  k_seg<<<kB, 256, 0, stream>>>(hard_ws, seg_ex, seg_in, cnt, possum, posds);

  // downsample: x_ds = [null; segment means]
  k_ds_init<<<dim3(kS1, kB), 256, 0, stream>>>(null_group, xds);
  k_ds_acc<<<dim3(kT, kB), 256, 0, stream>>>(xbuf, seg_ex, cnt, xds);
  hipMemcpyAsync(yds, xds, bufA, hipMemcpyDeviceToDevice, stream);

  run_layer(2, yds, kS1, 4 * kD, posds, kS1 + 1); // mid layers (full causal)
  run_layer(3, yds, kS1, 4 * kD, posds, kS1 + 1);

  // upsample + residual into xbuf
  k_upsample<<<dim3(kT, kB), 256, 0, stream>>>(xbuf, yds, xds, seg_in);

  run_layer(4, xbuf, kT, kD, nullptr, kWIN);      // up layers
  run_layer(5, xbuf, kT, kD, nullptr, kWIN);

  // logits + log_softmax
  gemm(xbuf, out_w, out_b, out_logits, kB * kT, kV, kD, GF_BIAS);
  k_logsoftmax<<<kB * kT, 256, 0, stream>>>(out_logits, kV);
}
